// FrameTransformer_14362370638495
// MI455X (gfx1250) — matrix operand order repaired and checked by reading, not compiled, not run on hardware
//
#include <hip/hip_runtime.h>
#include <hip/hip_bf16.h>

typedef __bf16 bf16;
typedef __attribute__((ext_vector_type(16))) __bf16 v16bf;
typedef __attribute__((ext_vector_type(8)))  __bf16 v8bf;
typedef __attribute__((ext_vector_type(8)))  float  v8f;

union ABfrag { v16bf v; v8bf h[2]; bf16 e[16]; };

#define FLAG_BIAS 1
#define FLAG_RELU 2
#define FLAG_ACC  4
#define FLAG_OBF  8

// ---------------------------------------------------------------------------
// NT WMMA GEMM:  C[m,n] = sum_k A[m,k] * B[n,k]   (B row-major [N][K])
// 256 threads = 8 waves. Block tile 128x128; each wave owns 16 rows x 128 cols
// (8 f32 accumulators). B tile (128 cols x 64 K = 16KB) staged cooperatively
// through LDS, double buffered (32KB) -> one barrier per 64-K stage, and the
// 8 waves share a single copy of B. All 8 B fragments are preloaded into
// registers before the WMMA burst so ds_load latency overlaps WMMA issue.
// A rows outside a conv segment (shiftA/segMask) are redirected to a zeroed
// dummy row -> branch-free inner loop, EXEC stays all-ones for WMMA.
// z-batching: offsets = (z>>3)*sHi + (z&7)*sLo.
// ---------------------------------------------------------------------------
template<int FLAGS>
__global__ __launch_bounds__(256)
void k_gemm_nt(const bf16* __restrict__ A, int lda, long long sAhi, long long sAlo,
               const bf16* __restrict__ Bm, int ldb, long long sBhi, long long sBlo,
               float* __restrict__ Cf, bf16* __restrict__ Cb, int ldc,
               long long sChi, long long sClo,
               const float* __restrict__ bias, int K, int shiftA, int segMask,
               const bf16* __restrict__ zrow)
{
    __shared__ __align__(32) bf16 sB[2][128 * 64];   // 32 KB double buffer

    const int lane = threadIdx.x & 31;
    const int wave = threadIdx.x >> 5;
    const int hi   = lane >> 4;
    const int l15  = lane & 15;
    const int zh = blockIdx.z >> 3, zl = blockIdx.z & 7;
    const long long offA = (long long)zh * sAhi + (long long)zl * sAlo;
    const long long offB = (long long)zh * sBhi + (long long)zl * sBlo;
    const long long offC = (long long)zh * sChi + (long long)zl * sClo;
    const int rm = blockIdx.y * 128 + wave * 16;
    const int cn = blockIdx.x * 128;

    // loop-invariant A row selection (invalid conv rows -> zeroed dummy row)
    const int  mrow   = rm + l15;
    const int  srcRow = mrow + shiftA;
    const bool valid  = ((unsigned)((mrow & segMask) + shiftA) < (unsigned)(segMask + 1));
    const bf16* aptr  = valid ? (A + offA + (long long)srcRow * lda) : zrow;

    // cooperative B loader: 256 threads x 64B = 16KB tile [col][k0..k63]
    const int bcol  = threadIdx.x >> 1;
    const int bhalf = (threadIdx.x & 1) * 32;
    const bf16* bsrc0 = Bm + offB + (long long)(cn + bcol) * ldb + bhalf;
    bf16* bdst0 = &sB[0][0] + bcol * 64 + bhalf;
    auto loadB = [&](int kb, int p) {
        const bf16* src = bsrc0 + kb;
        v8bf* dst = (v8bf*)(bdst0 + p * (128 * 64));
#pragma unroll
        for (int q = 0; q < 4; ++q) dst[q] = *(const v8bf*)(src + q * 8);
        __builtin_prefetch((const void*)(src + 128), 0, 3);  // global_prefetch_b8, WGP scope
    };

    v8f acc[8];
#pragma unroll
    for (int i = 0; i < 8; ++i) acc[i] = (v8f){0.f,0.f,0.f,0.f,0.f,0.f,0.f,0.f};

    loadB(0, 0);
    __syncthreads();
    int p = 0;
    for (int kb = 0; kb < K; kb += 64) {
        if (kb + 64 < K) loadB(kb + 64, p ^ 1);
        __builtin_prefetch((const void*)(aptr + kb + 128), 0, 3);

#pragma unroll
        for (int s = 0; s < 2; ++s) {
            // A fragment: slots 0..15 -> K=hi*16+j (matches B)
            ABfrag a;
            const bf16* ar = aptr + kb + s * 32;
            a.h[0] = *(const v8bf*)(ar + hi * 16);
            a.h[1] = *(const v8bf*)(ar + hi * 16 + 8);

            // preload all 8 B fragments, then burst 8 WMMAs
            ABfrag bfr[8];
#pragma unroll
            for (int i = 0; i < 8; ++i) {
                const bf16* bs = &sB[p][(i * 16 + l15) * 64 + s * 32 + hi * 16];
                bfr[i].h[0] = *(const v8bf*)(bs);
                bfr[i].h[1] = *(const v8bf*)(bs + 8);
            }
#pragma unroll
            for (int i = 0; i < 8; ++i)
                acc[i] = __builtin_amdgcn_wmma_f32_16x16x32_bf16(
                    false, a.v, false, bfr[i].v, (short)0, acc[i], false, false);
        }
        __syncthreads();
        p ^= 1;
    }

    // epilogue: C/D layout: VGPR r, lanes 0-15 -> M=r, lanes 16-31 -> M=8+r
#pragma unroll
    for (int i = 0; i < 8; ++i) {
        const int col = cn + i * 16 + l15;
#pragma unroll
        for (int r = 0; r < 8; ++r) {
            const int row = rm + hi * 8 + r;
            float v = acc[i][r];
            if (FLAGS & FLAG_BIAS) v += bias[col];
            if (FLAGS & FLAG_RELU) v = fmaxf(v, 0.f);
            const long long idx = offC + (long long)row * ldc + col;
            if (FLAGS & FLAG_OBF)      Cb[idx] = (bf16)v;
            else if (FLAGS & FLAG_ACC) Cf[idx] += v;
            else                       Cf[idx] = v;
        }
    }
}

// ---------------------------------------------------------------------------
// NN WMMA GEMM: C[m,n] = sum_k A[m,k] * B[k,n]  (B row-major [K][N]).
// Used only for q@er (K=128) and attn@V (K=512) -- ~4% of total FLOPs.
// 128 threads = 4 waves, 64x64 tile, strided scalar B loads.
// ---------------------------------------------------------------------------
template<int FLAGS>
__global__ __launch_bounds__(128)
void k_gemm_nn(const bf16* __restrict__ A, int lda, long long sAhi, long long sAlo,
               const bf16* __restrict__ Bm, int ldb, long long sBhi, long long sBlo,
               float* __restrict__ Cf, bf16* __restrict__ Cb, int ldc,
               long long sChi, long long sClo,
               const float* __restrict__ bias, int K)
{
    const int lane = threadIdx.x & 31;
    const int wave = threadIdx.x >> 5;
    const int hi   = lane >> 4;
    const int l15  = lane & 15;
    const int zh = blockIdx.z >> 3, zl = blockIdx.z & 7;
    const long long offA = (long long)zh * sAhi + (long long)zl * sAlo;
    const long long offB = (long long)zh * sBhi + (long long)zl * sBlo;
    const long long offC = (long long)zh * sChi + (long long)zl * sClo;
    const int rm = blockIdx.y * 64 + wave * 16;
    const int cn = blockIdx.x * 64;
    const bf16* aptr = A + offA + (long long)(rm + l15) * lda;

    v8f acc[4];
#pragma unroll
    for (int i = 0; i < 4; ++i) acc[i] = (v8f){0.f,0.f,0.f,0.f,0.f,0.f,0.f,0.f};

    for (int kb = 0; kb < K; kb += 32) {
        ABfrag a;
        const bf16* ar = aptr + kb;
        a.h[0] = *(const v8bf*)(ar + hi * 16);
        a.h[1] = *(const v8bf*)(ar + hi * 16 + 8);

#pragma unroll
        for (int i = 0; i < 4; ++i) {
            const int n = cn + i * 16 + l15;
            ABfrag b;
#pragma unroll
            for (int j = 0; j < 16; ++j)
                b.e[j] = Bm[offB + (long long)(kb + 16 * hi + j) * ldb + n];
            acc[i] = __builtin_amdgcn_wmma_f32_16x16x32_bf16(
                false, a.v, false, b.v, (short)0, acc[i], false, false);
        }
    }

#pragma unroll
    for (int i = 0; i < 4; ++i) {
        const int col = cn + i * 16 + l15;
#pragma unroll
        for (int r = 0; r < 8; ++r) {
            const int row = rm + hi * 8 + r;
            float v = acc[i][r];
            if (FLAGS & FLAG_BIAS) v += bias[col];
            if (FLAGS & FLAG_RELU) v = fmaxf(v, 0.f);
            const long long idx = offC + (long long)row * ldc + col;
            if (FLAGS & FLAG_OBF)      Cb[idx] = (bf16)v;
            else if (FLAGS & FLAG_ACC) Cf[idx] += v;
            else                       Cf[idx] = v;
        }
    }
}

// ---------------------------------------------------------------------------
// Elementwise / normalization kernels
// ---------------------------------------------------------------------------

__global__ void k_bnprep(const float* g, const float* bta, const float* mean,
                         const float* var, const float* wb, float* out)
{
    __shared__ float sadd[16];
    int c = threadIdx.x;
    if (c < 16) {
        float rs = rsqrtf(var[c] + 1e-5f);
        float sc = rs * g[c];
        out[c]   = sc * wb[c];
        sadd[c]  = (bta[c] - mean[c] * sc) * wb[c];
    }
    __syncthreads();
    if (c == 0) { float s = 0.f; for (int i = 0; i < 16; ++i) s += sadd[i]; out[16] = s; }
}

// xs[b,w,h] = relu( sum_c x[b,c,h,w]*scale[c] + addc ), LDS tile transpose so
// both the (w-contiguous) reads and (h-contiguous) writes are coalesced.
__global__ __launch_bounds__(256)
void k_bottleneck(const float* __restrict__ x, const float* __restrict__ sc,
                  float* __restrict__ xs)
{
    __shared__ float tile[32][33];
    const int b  = blockIdx.z;
    const int h0 = blockIdx.y * 32;
    const int w0 = blockIdx.x * 32;
    const float addc = sc[16];
    const int wl = threadIdx.x & 31;
    const int hg = threadIdx.x >> 5;          // 0..7
#pragma unroll
    for (int i = 0; i < 4; ++i) {
        const int hl = i * 8 + hg;
        const float* xp = x + (((long long)b * 16) * 1024 + (h0 + hl)) * 512 + w0 + wl;
        float acc = addc;
#pragma unroll
        for (int c = 0; c < 16; ++c) acc += xp[(long long)c * 524288] * sc[c];
        tile[hl][wl] = fmaxf(acc, 0.f);
    }
    __syncthreads();
    const int hl2 = threadIdx.x & 31;
    const int wg  = threadIdx.x >> 5;
#pragma unroll
    for (int i = 0; i < 4; ++i) {
        const int wl2 = i * 8 + wg;
        xs[((long long)b * 512 + w0 + wl2) * 1024 + h0 + hl2] = tile[hl2][wl2];
    }
}

template<bool WF, bool WB>
__global__ __launch_bounds__(256)
void k_layernorm(const float* __restrict__ in, float* __restrict__ outF,
                 bf16* __restrict__ outB, const float* __restrict__ g,
                 const float* __restrict__ bb, int D)
{
    __shared__ float red[256];
    const int tid = threadIdx.x;
    const long long base = (long long)blockIdx.x * D;
    float s = 0.f;
    for (int i = tid; i < D; i += 256) s += in[base + i];
    red[tid] = s; __syncthreads();
    for (int st = 128; st > 0; st >>= 1) { if (tid < st) red[tid] += red[tid + st]; __syncthreads(); }
    const float mean = red[0] * (1.f / D);
    __syncthreads();
    float s2 = 0.f;
    for (int i = tid; i < D; i += 256) { float d = in[base + i] - mean; s2 += d * d; }
    red[tid] = s2; __syncthreads();
    for (int st = 128; st > 0; st >>= 1) { if (tid < st) red[tid] += red[tid + st]; __syncthreads(); }
    const float rstd = rsqrtf(red[0] * (1.f / D) + 1e-5f);
    for (int i = tid; i < D; i += 256) {
        float v = (in[base + i] - mean) * rstd * g[i] + bb[i];
        if (WF) outF[base + i] = v;
        if (WB) outB[base + i] = (bf16)v;
    }
}

__global__ void k_glu(float* __restrict__ xs, const float* __restrict__ vv,
                      const float* __restrict__ gg)
{
    long long i = (long long)blockIdx.x * 256 + threadIdx.x;
    xs[i] += vv[i] * (1.f / (1.f + __expf(-gg[i])));
}

// hcat = relu(hL) + pad(relu(hR)) , in-place on t1 [T,4096]
__global__ void k_hcat(float* __restrict__ t1, const float* __restrict__ t2)
{
    long long i = (long long)blockIdx.x * 256 + threadIdx.x;
    int c = (int)(i & 4095);
    float v = fmaxf(t1[i], 0.f);
    if (c < 1024) v += fmaxf(t2[(i >> 12) * 1024 + c], 0.f);
    t1[i] = v;
}

// depthwise conv k=9 over w, [b,w,c] layout, bf16 output for the pw GEMM
__global__ void k_dwconv(const float* __restrict__ in, const float* __restrict__ wdw,
                         bf16* __restrict__ out)
{
    long long i = (long long)blockIdx.x * 256 + threadIdx.x;   // T*4096
    int c = (int)(i & 4095);
    long long t = i >> 12;
    int w = (int)(t & 511);
    float acc = 0.f;
#pragma unroll
    for (int j = 0; j < 9; ++j) {
        int wsrc = w + j - 4;
        if ((unsigned)wsrc < 512u) acc += in[(t + j - 4) * 4096 + c] * wdw[c * 9 + j];
    }
    out[i] = (bf16)acc;
}

// sa[z,i,j] = (sa[z,i,j] + qe[z,j,i]) / 32   (relative-position "skew")
__global__ void k_skew(float* __restrict__ sa, const float* __restrict__ qe)
{
    long long i = (long long)blockIdx.x * 256 + threadIdx.x;   // 64*512*512
    int j  = (int)(i & 511);
    int ii = (int)((i >> 9) & 511);
    long long z = i >> 18;
    sa[i] = (sa[i] + qe[(z << 18) + ((long long)j << 9) + ii]) * 0.03125f;
}

__global__ __launch_bounds__(256)
void k_softmax(const float* __restrict__ sa, bf16* __restrict__ attn)
{
    __shared__ float red[256];
    const int tid = threadIdx.x;
    const long long base = (long long)blockIdx.x << 9;   // rows of 512
    float a0 = sa[base + tid], a1 = sa[base + 256 + tid];
    red[tid] = fmaxf(a0, a1); __syncthreads();
    for (int st = 128; st > 0; st >>= 1) { if (tid < st) red[tid] = fmaxf(red[tid], red[tid + st]); __syncthreads(); }
    const float mx = red[0];
    __syncthreads();
    float e0 = __expf(a0 - mx), e1 = __expf(a1 - mx);
    red[tid] = e0 + e1; __syncthreads();
    for (int st = 128; st > 0; st >>= 1) { if (tid < st) red[tid] += red[tid + st]; __syncthreads(); }
    const float inv = 1.f / red[0];
    attn[base + tid]       = (bf16)(e0 * inv);
    attn[base + 256 + tid] = (bf16)(e1 * inv);
}

__global__ void k_resid(float* __restrict__ xs, const float* __restrict__ h)
{
    long long i = (long long)blockIdx.x * 256 + threadIdx.x;
    xs[i] += h[i];
}

// out[b,0,h,w] = xs[b,w,h] with LDS tile transpose (coalesced both sides)
__global__ __launch_bounds__(256)
void k_outT(const float* __restrict__ xs, float* __restrict__ out)
{
    __shared__ float tile[32][33];
    const int b  = blockIdx.z;
    const int h0 = blockIdx.y * 32;
    const int w0 = blockIdx.x * 32;
    const int hl = threadIdx.x & 31;
    const int wg = threadIdx.x >> 5;
#pragma unroll
    for (int i = 0; i < 4; ++i) {
        const int wl = i * 8 + wg;
        tile[wl][hl] = xs[((long long)b * 512 + w0 + wl) * 1024 + h0 + hl];
    }
    __syncthreads();
    const int wl2 = threadIdx.x & 31;
    const int hg  = threadIdx.x >> 5;
#pragma unroll
    for (int i = 0; i < 4; ++i) {
        const int h = h0 + i * 8 + hg;
        out[((long long)b * 1024 + h) * 512 + w0 + wl2] = tile[wl2][i * 8 + hg];
    }
}

// strided f32 -> bf16 convert (stride/off splits conv1R [O][I][3] slices)
__global__ void k_cvt(const float* __restrict__ s, bf16* __restrict__ d,
                      long long n, int stride, int off)
{
    long long i = (long long)blockIdx.x * 256 + threadIdx.x;
    if (i < n) d[i] = (bf16)s[i * (long long)stride + off];
}

__global__ void k_zero_bf16(bf16* __restrict__ d)
{
    d[blockIdx.x * 256 + threadIdx.x] = (bf16)0.0f;
}

// ---------------------------------------------------------------------------
extern "C" void kernel_launch(void* const* d_in, const int* in_sizes, int n_in,
                              void* d_out, int out_size, void* d_ws, size_t ws_size,
                              hipStream_t stream)
{
    (void)in_sizes; (void)n_in; (void)out_size; (void)ws_size;
    const int T    = 4096;          // 8 batches * 512 frames
    const int BIGM = 0x3FFFFFFF;    // "no segment" mask

    const float* x     = (const float*)d_in[0];
    const float* bng   = (const float*)d_in[1];
    const float* bnb   = (const float*)d_in[2];
    const float* bnm   = (const float*)d_in[3];
    const float* bnv   = (const float*)d_in[4];
    const float* wbk   = (const float*)d_in[5];
    const float* ln1g  = (const float*)d_in[6];  const float* ln1b = (const float*)d_in[7];
    const float* wgluv = (const float*)d_in[8];
    const float* wglug = (const float*)d_in[9];
    const float* ln2g  = (const float*)d_in[10]; const float* ln2b = (const float*)d_in[11];
    const float* w1L   = (const float*)d_in[12];
    const float* w1R   = (const float*)d_in[13];
    const float* ln3g  = (const float*)d_in[14]; const float* ln3b = (const float*)d_in[15];
    const float* wdw   = (const float*)d_in[16];
    const float* wpw   = (const float*)d_in[17];
    const float* ln4g  = (const float*)d_in[18]; const float* ln4b = (const float*)d_in[19];
    const float* wq    = (const float*)d_in[20]; const float* bq   = (const float*)d_in[21];
    const float* wk    = (const float*)d_in[22]; const float* bk   = (const float*)d_in[23];
    const float* wv    = (const float*)d_in[24]; const float* bv   = (const float*)d_in[25];
    const float* wo    = (const float*)d_in[26]; const float* bo   = (const float*)d_in[27];
    const float* er    = (const float*)d_in[28];
    const float* ln5g  = (const float*)d_in[29]; const float* ln5b = (const float*)d_in[30];
    const float* w3    = (const float*)d_in[31];
    const float* w4    = (const float*)d_in[32];

    float* outMain = (float*)d_out;             // [8,1,1024,512]
    float* sa      = (float*)d_out + 4194304;   // [8,8,512,512] logits

    // ---- scratch layout ----
    char* wsb = (char*)d_ws;
    size_t off = 0;
    auto alloc = [&](size_t bytes) -> void* {
        void* p = wsb + off; off += (bytes + 255) & ~(size_t)255; return p;
    };
    float* xs    = (float*)alloc((size_t)T * 1024 * 4);
    float* t1    = (float*)alloc((size_t)T * 4096 * 4);
    float* t2    = (float*)alloc((size_t)T * 4096 * 4);
    bf16*  abf   = (bf16*) alloc((size_t)T * 4096 * 2);
    bf16*  abf2  = (bf16*) alloc((size_t)T * 4096 * 2);
    float* qeb   = (float*)alloc((size_t)64 * 512 * 512 * 4);
    bf16*  qb    = (bf16*) alloc((size_t)T * 1024 * 2);
    bf16*  kbuf  = (bf16*) alloc((size_t)T * 1024 * 2);
    bf16*  vbuf  = (bf16*) alloc((size_t)T * 1024 * 2);
    bf16*  attnb = (bf16*) alloc((size_t)64 * 512 * 512 * 2);
    float* bnsc  = (float*)alloc(64 * 4);
    bf16*  zrow  = (bf16*) alloc((size_t)4096 * 2);
    bf16* wgvB = (bf16*)alloc((size_t)1024 * 1024 * 2);
    bf16* wggB = (bf16*)alloc((size_t)1024 * 1024 * 2);
    bf16* w1LB = (bf16*)alloc((size_t)4096 * 1024 * 2);
    bf16* w1R0 = (bf16*)alloc((size_t)1024 * 1024 * 2);
    bf16* w1R1 = (bf16*)alloc((size_t)1024 * 1024 * 2);
    bf16* w1R2 = (bf16*)alloc((size_t)1024 * 1024 * 2);
    bf16* wpwB = (bf16*)alloc((size_t)1024 * 4096 * 2);
    bf16* wqB  = (bf16*)alloc((size_t)1024 * 1024 * 2);
    bf16* wkB  = (bf16*)alloc((size_t)1024 * 1024 * 2);
    bf16* wvB  = (bf16*)alloc((size_t)1024 * 1024 * 2);
    bf16* woB  = (bf16*)alloc((size_t)1024 * 1024 * 2);
    bf16* erB  = (bf16*)alloc((size_t)128 * 512 * 2);
    bf16* w3B  = (bf16*)alloc((size_t)4096 * 1024 * 2);
    bf16* w4B  = (bf16*)alloc((size_t)1024 * 4096 * 2);

    auto cvt = [&](const float* s, bf16* d, long long n, int stride, int o) {
        k_cvt<<<dim3((unsigned)((n + 255) / 256)), 256, 0, stream>>>(s, d, n, stride, o);
    };

    // ---- weight prep ----
    k_bnprep<<<1, 32, 0, stream>>>(bng, bnb, bnm, bnv, wbk, bnsc);
    k_zero_bf16<<<16, 256, 0, stream>>>(zrow);
    cvt(wgluv, wgvB, 1048576, 1, 0);
    cvt(wglug, wggB, 1048576, 1, 0);
    cvt(w1L,   w1LB, 4194304, 1, 0);
    cvt(w1R,   w1R0, 1048576, 3, 0);
    cvt(w1R,   w1R1, 1048576, 3, 1);
    cvt(w1R,   w1R2, 1048576, 3, 2);
    cvt(wpw,   wpwB, 4194304, 1, 0);
    cvt(wq,    wqB,  1048576, 1, 0);
    cvt(wk,    wkB,  1048576, 1, 0);
    cvt(wv,    wvB,  1048576, 1, 0);
    cvt(wo,    woB,  1048576, 1, 0);
    cvt(er,    erB,  65536,   1, 0);
    cvt(w3,    w3B,  4194304, 1, 0);
    cvt(w4,    w4B,  4194304, 1, 0);

    const dim3 blkNT(256), blkNN(128);
    const dim3 gM_1024(8, 32, 1);    // N=1024, M=4096 (128x128 tiles)
    const dim3 gM_4096(32, 32, 1);   // N=4096, M=4096
    const dim3 gTrans(16, 32, 8);    // 32x32 transpose tiles
    const unsigned ew1024 = (unsigned)((long long)T * 1024 / 256);
    const unsigned ew4096 = (unsigned)((long long)T * 4096 / 256);
    const unsigned ewAttn = (unsigned)((long long)64 * 512 * 512 / 256);

    // ---- stage 1: BN + bottleneck + relu ----
    k_bottleneck<<<gTrans, 256, 0, stream>>>(x, bnsc, xs);

    // ---- stage 2: GLU block ----
    k_layernorm<false, true><<<T, 256, 0, stream>>>(xs, nullptr, abf, ln1g, ln1b, 1024);
    k_gemm_nt<0><<<gM_1024, blkNT, 0, stream>>>(abf, 1024, 0, 0, wgvB, 1024, 0, 0,
        t1, nullptr, 1024, 0, 0, nullptr, 1024, 0, BIGM, zrow);
    k_gemm_nt<0><<<gM_1024, blkNT, 0, stream>>>(abf, 1024, 0, 0, wggB, 1024, 0, 0,
        t2, nullptr, 1024, 0, 0, nullptr, 1024, 0, BIGM, zrow);
    k_glu<<<ew1024, 256, 0, stream>>>(xs, t1, t2);

    // ---- stage 3: conv feedforward block ----
    k_layernorm<false, true><<<T, 256, 0, stream>>>(xs, nullptr, abf, ln2g, ln2b, 1024);
    k_gemm_nt<0><<<gM_4096, blkNT, 0, stream>>>(abf, 1024, 0, 0, w1LB, 1024, 0, 0,
        t1, nullptr, 4096, 0, 0, nullptr, 1024, 0, BIGM, zrow);
    // conv1d k=3 over w as 3 shifted accumulated GEMMs (per-batch zero padding)
    k_gemm_nt<0><<<gM_1024, blkNT, 0, stream>>>(abf, 1024, 0, 0, w1R0, 1024, 0, 0,
        t2, nullptr, 1024, 0, 0, nullptr, 1024, -1, 511, zrow);
    k_gemm_nt<FLAG_ACC><<<gM_1024, blkNT, 0, stream>>>(abf, 1024, 0, 0, w1R1, 1024, 0, 0,
        t2, nullptr, 1024, 0, 0, nullptr, 1024, 0, 511, zrow);
    k_gemm_nt<FLAG_ACC><<<gM_1024, blkNT, 0, stream>>>(abf, 1024, 0, 0, w1R2, 1024, 0, 0,
        t2, nullptr, 1024, 0, 0, nullptr, 1024, 1, 511, zrow);
    k_hcat<<<ew4096, 256, 0, stream>>>(t1, t2);
    k_layernorm<true, false><<<T, 256, 0, stream>>>(t1, t2, nullptr, ln3g, ln3b, 4096);
    k_dwconv<<<ew4096, 256, 0, stream>>>(t2, wdw, abf);
    k_gemm_nt<0><<<gM_1024, blkNT, 0, stream>>>(abf, 4096, 0, 0, wpwB, 4096, 0, 0,
        t2, nullptr, 1024, 0, 0, nullptr, 4096, 0, BIGM, zrow);
    k_resid<<<ew1024, 256, 0, stream>>>(xs, t2);

    // ---- stage 4: multiband relative-position attention ----
    k_layernorm<false, true><<<T, 256, 0, stream>>>(xs, nullptr, abf, ln4g, ln4b, 1024);
    k_gemm_nt<FLAG_BIAS | FLAG_OBF><<<gM_1024, blkNT, 0, stream>>>(abf, 1024, 0, 0,
        wqB, 1024, 0, 0, nullptr, qb, 1024, 0, 0, bq, 1024, 0, BIGM, zrow);
    k_gemm_nt<FLAG_BIAS | FLAG_OBF><<<gM_1024, blkNT, 0, stream>>>(abf, 1024, 0, 0,
        wkB, 1024, 0, 0, nullptr, kbuf, 1024, 0, 0, bk, 1024, 0, BIGM, zrow);
    k_gemm_nt<FLAG_BIAS | FLAG_OBF><<<gM_1024, blkNT, 0, stream>>>(abf, 1024, 0, 0,
        wvB, 1024, 0, 0, nullptr, vbuf, 1024, 0, 0, bv, 1024, 0, BIGM, zrow);

    // qe[b,nb] = q_band [512,128] @ er [128,512]   (NN, z = b*8+nb)
    k_gemm_nn<0><<<dim3(8, 8, 64), blkNN, 0, stream>>>(
        qb, 1024, 524288, 128, erB, 512, 0, 0,
        qeb, nullptr, 512, 2097152, 262144, nullptr, 128);
    // logits = q_band @ k_band^T   (NT, batched) -> straight into sa output
    k_gemm_nt<0><<<dim3(4, 4, 64), blkNT, 0, stream>>>(
        qb, 1024, 524288, 128, kbuf, 1024, 524288, 128,
        sa, nullptr, 512, 2097152, 262144, nullptr, 128, 0, BIGM, zrow);
    k_skew<<<ewAttn, 256, 0, stream>>>(sa, qeb);
    k_softmax<<<64 * 512, 256, 0, stream>>>(sa, attnb);
    // o_band = attn [512,512] @ v_band [512,128]  (NN) -> bf16 into abf
    k_gemm_nn<FLAG_OBF><<<dim3(2, 8, 64), blkNN, 0, stream>>>(
        attnb, 512, 2097152, 262144, vbuf, 1024, 524288, 128,
        nullptr, abf, 1024, 524288, 128, nullptr, 512);
    k_gemm_nt<FLAG_BIAS><<<gM_1024, blkNT, 0, stream>>>(abf, 1024, 0, 0,
        woB, 1024, 0, 0, t1, nullptr, 1024, 0, 0, bo, 1024, 0, BIGM, zrow);
    k_resid<<<ew1024, 256, 0, stream>>>(xs, t1);

    // ---- stage 5: final MLP ----
    k_layernorm<false, true><<<T, 256, 0, stream>>>(xs, nullptr, abf, ln5g, ln5b, 1024);
    k_gemm_nt<FLAG_RELU | FLAG_OBF><<<gM_4096, blkNT, 0, stream>>>(abf, 1024, 0, 0,
        w3B, 1024, 0, 0, nullptr, abf2, 4096, 0, 0, nullptr, 1024, 0, BIGM, zrow);
    k_gemm_nt<0><<<gM_1024, blkNT, 0, stream>>>(abf2, 4096, 0, 0,
        w4B, 4096, 0, 0, t1, nullptr, 1024, 0, 0, nullptr, 4096, 0, BIGM, zrow);
    k_resid<<<ew1024, 256, 0, stream>>>(xs, t1);

    // ---- output transpose ----
    k_outT<<<gTrans, 256, 0, stream>>>(xs, outMain);
}
